// StrokeARFMv41_12317966205185
// MI455X (gfx1250) — compile-verified, matrix operand order checked by reading, not run on hardware
//
#include <hip/hip_runtime.h>
#include <hip/hip_bf16.h>
#include <math.h>

#define DEV __device__ __forceinline__

typedef __attribute__((ext_vector_type(8)))  float  v8f;
typedef __attribute__((ext_vector_type(8)))  __bf16 v8bf;
typedef __attribute__((ext_vector_type(16))) __bf16 v16bf;

// ---------------- model dims ----------------
constexpr int B_    = 512;
constexpr int T_    = 50;
constexpr int P_    = 196;
constexpr int PAST_ = 50;
constexpr int D_    = 256;
constexpr int H_    = 8;
constexpr int DH_   = 32;
constexpr int FF_   = 1024;
constexpr int L_    = 6;

// ---------------- helpers ----------------
DEV unsigned f2bf(float x) {
  unsigned u = __float_as_uint(x);
  return (u + 0x7FFFu + ((u >> 16) & 1u)) >> 16;  // RNE, low 16 bits valid
}

// Load a 16x32 bf16 A/B fragment from an LDS tile (row-major, strideE halves/row).
// ISA 7.12.2 16-bit layout: lane holds row M = lane%16; half = lane/16 selects the
// K subgroups -> two contiguous 16B chunks at column offsets half*8 and 16+half*8.
DEV v16bf frag_ld(const unsigned short* t, int row, int strideE, int kOff, int lane) {
  int half = (lane >> 4) & 1;
  const unsigned short* p = t + row * strideE + kOff + half * 8;
  v8bf lo = *(const v8bf*)(p);
  v8bf hi = *(const v8bf*)(p + 16);
  return __builtin_shufflevector(lo, hi, 0,1,2,3,4,5,6,7,8,9,10,11,12,13,14,15);
}

DEV v8f wmma_bf16(v16bf a, v16bf b, v8f c) {
  return __builtin_amdgcn_wmma_f32_16x16x32_bf16(false, a, false, b, (short)0, c, false, false);
}

// CDNA5 async copy global->LDS (ASYNCcnt); one b128 per lane.
DEV void async_ld_b128(void* lds_dst, const float* g) {
  unsigned loff = (unsigned)(unsigned long long)(uintptr_t)lds_dst;  // low 32b = LDS offset
  asm volatile("global_load_async_to_lds_b128 %0, %1, off"
               :: "v"(loff), "v"((unsigned long long)(uintptr_t)g)
               : "memory");
}
DEV void wait_async0() { asm volatile("s_wait_asynccnt 0x0" ::: "memory"); }

// ---------------- generic GEMM:  C = act(A[MxK] . W[NxK]^T + bias) (+ add) ----------------
// Block tile 128x128, BK=32, 8 wave32 waves in a 4(M) x 2(N) grid -> 32x64 per wave,
// 8 WMMAs per wave per K-step. Interior tiles: async double-buffered fp32 staging
// (global_load_async_to_lds_b128) + vectorized fp32->bf16 convert; edge tiles: guarded path.
constexpr int GBM = 128, GBN = 128, GBK = 32;
constexpr int GST = 40;  // bf16 tile stride (halves), padded to dodge bank conflicts

__global__ __launch_bounds__(256) void gemm_bf16_kernel(
    const float* __restrict__ A, const float* __restrict__ W,
    const float* __restrict__ bias, const float* __restrict__ add,
    float* __restrict__ C, int M, int N, int K, int act)
{
  __shared__ float  Afs[2][GBM * GBK];   // raw fp32 double buffers (async dst)
  __shared__ float  Wfs[2][GBN * GBK];
  __shared__ unsigned short As[GBM * GST];  // bf16 tiles consumed by WMMA
  __shared__ unsigned short Ws[GBN * GST];

  int tid = threadIdx.x, lane = tid & 31, wid = tid >> 5;
  int wm = wid >> 1, wn = wid & 1;
  int m0 = blockIdx.y * GBM, n0 = blockIdx.x * GBN;
  int half = lane >> 4, nl = lane & 15;
  int nk = (K + GBK - 1) / GBK;

  v8f acc[2][4];
  #pragma unroll
  for (int i = 0; i < 2; ++i)
    #pragma unroll
    for (int j = 0; j < 4; ++j)
      #pragma unroll
      for (int e = 0; e < 8; ++e) acc[i][j][e] = 0.f;

  bool fast = (m0 + GBM <= M) && (n0 + GBN <= N) && ((K & (GBK - 1)) == 0);

  if (fast) {
    // ---- async double-buffered pipeline ----
    // per-thread copy slots: 4 x b128 for A tile, 4 x b128 for W tile
    int rowA = tid >> 3, c4 = (tid & 7) * 4;          // 32 rows per 256-thread pass? no:
    // e = tid + i*256 -> row = e>>3 (0..127), col4 = (e&7)*4
    #pragma unroll
    for (int i = 0; i < 4; ++i) {                     // prologue: tile 0
      int e = tid + i * 256, r = e >> 3, c = (e & 7) * 4;
      async_ld_b128(&Afs[0][r * GBK + c], A + (size_t)(m0 + r) * K + c);
      async_ld_b128(&Wfs[0][r * GBK + c], W + (size_t)(n0 + r) * K + c);
    }
    (void)rowA; (void)c4;
    for (int kt = 0; kt < nk; ++kt) {
      int cur = kt & 1;
      wait_async0();
      __syncthreads();                                 // fp32 tile `cur` ready everywhere
      if (kt + 1 < nk) {
        int k0 = (kt + 1) * GBK;
        #pragma unroll
        for (int i = 0; i < 4; ++i) {
          int e = tid + i * 256, r = e >> 3, c = (e & 7) * 4;
          async_ld_b128(&Afs[cur ^ 1][r * GBK + c], A + (size_t)(m0 + r) * K + k0 + c);
          async_ld_b128(&Wfs[cur ^ 1][r * GBK + c], W + (size_t)(n0 + r) * K + k0 + c);
        }
      }
      // convert fp32 -> packed bf16 (b128 LDS reads, b64 LDS writes)
      #pragma unroll
      for (int i = 0; i < 4; ++i) {
        int e = tid + i * 256, r = e >> 3, c = (e & 7) * 4;
        const float* s = &Afs[cur][r * GBK + c];
        unsigned lo = f2bf(s[0]) | (f2bf(s[1]) << 16);
        unsigned hi = f2bf(s[2]) | (f2bf(s[3]) << 16);
        unsigned* d = (unsigned*)&As[r * GST + c];
        d[0] = lo; d[1] = hi;
        const float* sw = &Wfs[cur][r * GBK + c];
        unsigned lw = f2bf(sw[0]) | (f2bf(sw[1]) << 16);
        unsigned hw = f2bf(sw[2]) | (f2bf(sw[3]) << 16);
        unsigned* dw = (unsigned*)&Ws[r * GST + c];
        dw[0] = lw; dw[1] = hw;
      }
      __syncthreads();
      v16bf a0 = frag_ld(As, wm * 32 +  0 + nl, GST, 0, lane);
      v16bf a1 = frag_ld(As, wm * 32 + 16 + nl, GST, 0, lane);
      #pragma unroll
      for (int j = 0; j < 4; ++j) {
        v16bf bj = frag_ld(Ws, wn * 64 + j * 16 + nl, GST, 0, lane);
        acc[0][j] = wmma_bf16(a0, bj, acc[0][j]);
        acc[1][j] = wmma_bf16(a1, bj, acc[1][j]);
      }
    }
  } else {
    // ---- guarded fallback (only tiny head/edge GEMMs take this) ----
    for (int kt = 0; kt < nk; ++kt) {
      int k0 = kt * GBK;
      __syncthreads();
      for (int e = tid; e < GBM * GBK; e += 256) {
        int r = e >> 5, c = e & 31;
        int gm = m0 + r, gk = k0 + c;
        float v = (gm < M && gk < K) ? A[(size_t)gm * K + gk] : 0.f;
        As[r * GST + c] = (unsigned short)f2bf(v);
      }
      for (int e = tid; e < GBN * GBK; e += 256) {
        int r = e >> 5, c = e & 31;
        int gn = n0 + r, gk = k0 + c;
        float v = (gn < N && gk < K) ? W[(size_t)gn * K + gk] : 0.f;
        Ws[r * GST + c] = (unsigned short)f2bf(v);
      }
      __syncthreads();
      v16bf a0 = frag_ld(As, wm * 32 +  0 + nl, GST, 0, lane);
      v16bf a1 = frag_ld(As, wm * 32 + 16 + nl, GST, 0, lane);
      #pragma unroll
      for (int j = 0; j < 4; ++j) {
        v16bf bj = frag_ld(Ws, wn * 64 + j * 16 + nl, GST, 0, lane);
        acc[0][j] = wmma_bf16(a0, bj, acc[0][j]);
        acc[1][j] = wmma_bf16(a1, bj, acc[1][j]);
      }
    }
  }

  // store: D layout lane n = lane&15, row m = r + 8*(lane/16)
  #pragma unroll
  for (int mi = 0; mi < 2; ++mi)
    #pragma unroll
    for (int ni = 0; ni < 4; ++ni)
      #pragma unroll
      for (int r = 0; r < 8; ++r) {
        int gm = m0 + wm * 32 + mi * 16 + r + half * 8;
        int gn = n0 + wn * 64 + ni * 16 + nl;
        if (gm < M && gn < N) {
          float v = acc[mi][ni][r] + (bias ? bias[gn] : 0.f);
          if (act == 1) v = v / (1.f + __expf(-v));   // silu
          if (add) v += add[(size_t)gm * N + gn];
          C[(size_t)gm * N + gn] = v;
        }
      }
}

// ---------------- fused attention per (b,h): softmax(Q.K^T/sqrt(dh) + mask) . V ----------------
constexpr int AQST = 40;   // Q/K tile stride (halves)
constexpr int AVST = 264;  // V^T tile stride
constexpr int ASFS = 260;  // score f32 stride
constexpr int ASBS = 264;  // score bf16 stride

__global__ __launch_bounds__(256) void attn_kernel(
    const float* __restrict__ Q, const float* __restrict__ K,
    const float* __restrict__ V, float* __restrict__ O,
    int Tq, int Tk, int coff /* causal: mask j > coff+i, -1 = none */)
{
  __shared__ unsigned short Qs[64 * AQST];
  __shared__ unsigned short Ks[256 * AQST];
  __shared__ unsigned short Vt[32 * AVST];
  __shared__ float          Sf[64 * ASFS];
  __shared__ unsigned short Sb[64 * ASBS];

  int b = blockIdx.x / H_;
  int h = blockIdx.x % H_;
  int tid = threadIdx.x, lane = tid & 31, wid = tid >> 5;
  int half = lane >> 4, nl = lane & 15;
  int TKP = ((Tk + 31) / 32) * 32;   // padded key length (mult of 32)

  for (int e = tid; e < 64 * 32; e += 256) {
    int q = e >> 5, d = e & 31;
    float v = (q < Tq) ? Q[((size_t)b * Tq + q) * D_ + h * DH_ + d] : 0.f;
    Qs[q * AQST + d] = (unsigned short)f2bf(v);
  }
  for (int e = tid; e < 256 * 32; e += 256) {
    int tk = e >> 5, d = e & 31;
    float v = (tk < Tk) ? K[((size_t)b * Tk + tk) * D_ + h * DH_ + d] : 0.f;
    Ks[tk * AQST + d] = (unsigned short)f2bf(v);
  }
  for (int e = tid; e < 32 * 256; e += 256) {
    int d = e >> 8, tk = e & 255;
    float v = (tk < Tk) ? V[((size_t)b * Tk + tk) * D_ + h * DH_ + d] : 0.f;
    Vt[d * AVST + tk] = (unsigned short)f2bf(v);
  }
  __syncthreads();

  // scores: one WMMA per 16x16 tile (K-dim = DH = 32 exactly)
  int nkt = TKP / 16;
  for (int t5 = wid; t5 < 4 * nkt; t5 += 8) {
    int qt = t5 / nkt, kt = t5 % nkt;
    v16bf a  = frag_ld(Qs, qt * 16 + nl, AQST, 0, lane);
    v16bf bb = frag_ld(Ks, kt * 16 + nl, AQST, 0, lane);
    v8f s;
    #pragma unroll
    for (int e = 0; e < 8; ++e) s[e] = 0.f;
    s = wmma_bf16(a, bb, s);
    #pragma unroll
    for (int r = 0; r < 8; ++r) {
      int m = qt * 16 + r + half * 8;
      int n = kt * 16 + nl;
      float sv = s[r] * 0.17677669529663687f;  // 1/sqrt(32)
      bool msk = (n >= Tk) || (coff >= 0 && n > coff + m);
      Sf[m * ASFS + n] = msk ? -1e30f : sv;
    }
  }
  __syncthreads();

  // softmax rows (one wave per row, 8 rows/wave)
  for (int row = wid; row < 64; row += 8) {
    float mx = -3.0e38f;
    for (int c = lane; c < TKP; c += 32) mx = fmaxf(mx, Sf[row * ASFS + c]);
    for (int o = 16; o > 0; o >>= 1) mx = fmaxf(mx, __shfl_xor(mx, o, 32));
    float sum = 0.f;
    for (int c = lane; c < TKP; c += 32) {
      float e = __expf(Sf[row * ASFS + c] - mx);
      Sf[row * ASFS + c] = e;
      sum += e;
    }
    for (int o = 16; o > 0; o >>= 1) sum += __shfl_xor(sum, o, 32);
    float inv = 1.f / sum;
    for (int c = lane; c < TKP; c += 32)
      Sb[row * ASBS + c] = (unsigned short)f2bf(Sf[row * ASFS + c] * inv);
  }
  __syncthreads();

  // context: O[64x32] = Sb[64xTKP] . V[TKPx32]; 8 waves -> 4 qtiles x 2 dtiles
  int qt = wid >> 1, dt = wid & 1;
  v8f acc;
  #pragma unroll
  for (int e = 0; e < 8; ++e) acc[e] = 0.f;
  for (int kt = 0; kt < TKP / 32; ++kt) {
    v16bf a  = frag_ld(Sb, qt * 16 + nl, ASBS, kt * 32, lane);
    v16bf bb = frag_ld(Vt, dt * 16 + nl, AVST, kt * 32, lane);
    acc = wmma_bf16(a, bb, acc);
  }
  #pragma unroll
  for (int r = 0; r < 8; ++r) {
    int q = qt * 16 + r + half * 8;
    int d = dt * 16 + nl;
    if (q < Tq) O[((size_t)b * Tq + q) * D_ + h * DH_ + d] = acc[r];
  }
}

// ---------------- adaLN: out = (2 + scC_s + scT_s)*ln(x) + scC_sh + scT_sh ----------------
__global__ __launch_bounds__(256) void adaln_kernel(
    const float* __restrict__ x, const float* __restrict__ scC,
    const float* __restrict__ scT, float* __restrict__ out)
{
  int row = blockIdx.x;
  int d = threadIdx.x;
  float v = x[(size_t)row * D_ + d];
  float s = v, s2 = v * v;
  for (int o = 16; o > 0; o >>= 1) { s += __shfl_xor(s, o, 32); s2 += __shfl_xor(s2, o, 32); }
  __shared__ float ps[8], ps2[8];
  int lane = d & 31, wid = d >> 5;
  if (lane == 0) { ps[wid] = s; ps2[wid] = s2; }
  __syncthreads();
  float ts = 0.f, ts2 = 0.f;
  for (int k = 0; k < 8; ++k) { ts += ps[k]; ts2 += ps2[k]; }
  float mu = ts / 256.f;
  float var = ts2 / 256.f - mu * mu;
  float rs = rsqrtf(var + 1e-5f);
  float ln = (v - mu) * rs;
  int b = row / T_;
  float sc = scC[(size_t)b * 512 + d] + scT[(size_t)b * 512 + d];
  float sh = scC[(size_t)b * 512 + 256 + d] + scT[(size_t)b * 512 + 256 + d];
  out[(size_t)row * D_ + d] = (2.f + sc) * ln + sh;
}

// ---------------- LN affine ----------------
__global__ __launch_bounds__(256) void ln_affine_kernel(
    const float* __restrict__ x, const float* __restrict__ g, const float* __restrict__ bb,
    float* __restrict__ out, int rowOff, int rowStride)
{
  int rowIn = blockIdx.x * rowStride + rowOff;
  int d = threadIdx.x;
  float v = x[(size_t)rowIn * D_ + d];
  float s = v, s2 = v * v;
  for (int o = 16; o > 0; o >>= 1) { s += __shfl_xor(s, o, 32); s2 += __shfl_xor(s2, o, 32); }
  __shared__ float ps[8], ps2[8];
  int lane = d & 31, wid = d >> 5;
  if (lane == 0) { ps[wid] = s; ps2[wid] = s2; }
  __syncthreads();
  float ts = 0.f, ts2 = 0.f;
  for (int k = 0; k < 8; ++k) { ts += ps[k]; ts2 += ps2[k]; }
  float mu = ts / 256.f;
  float var = ts2 / 256.f - mu * mu;
  float rs = rsqrtf(var + 1e-5f);
  out[(size_t)blockIdx.x * D_ + d] = (v - mu) * rs * g[d] + bb[d];
}

// ---------------- misc elementwise ----------------
__global__ void add_pos_kernel(float* __restrict__ x, const float* __restrict__ pos, int n) {
  int i = blockIdx.x * 256 + threadIdx.x;
  if (i < n) x[i] += pos[i % (T_ * D_)];
}

__global__ void concat_kernel(const float* __restrict__ past, const float* __restrict__ cur,
                              float* __restrict__ dst, int n) {
  int i = blockIdx.x * 256 + threadIdx.x;
  if (i >= n) return;
  int d = i & (D_ - 1);
  int r = (i >> 8) % (PAST_ + T_);
  int b = i / ((PAST_ + T_) * D_);
  dst[i] = (r < PAST_) ? past[((size_t)b * PAST_ + r) * D_ + d]
                       : cur[((size_t)b * T_ + (r - PAST_)) * D_ + d];
}

__global__ void fm_build_kernel(const float* __restrict__ h, const float* __restrict__ t_rand,
                                const float* __restrict__ a_src, const float* __restrict__ a_tar,
                                float* __restrict__ fmin) {
  int i = blockIdx.x * 256 + threadIdx.x;
  if (i >= B_ * 296) return;
  int b = i / 296, c = i % 296;
  float out;
  if (c < 256) {
    out = h[(size_t)b * D_ + c];
  } else if (c < 264) {
    int j = c - 256;
    float t = t_rand[b];
    out = (1.f - t) * a_src[b * 8 + j] + t * a_tar[b * 8 + j];
  } else {
    int j = c - 264;
    int k = j & 15;
    float t = t_rand[b];
    float f = __expf(-logf(10000.f) * (float)k / 15.f);
    float arg = t * f;
    out = (j < 16) ? sinf(arg) : cosf(arg);
  }
  fmin[(size_t)b * 296 + c] = out;
}

__global__ void zero_kernel(float* o, int n) {
  int i = blockIdx.x * 256 + threadIdx.x;
  if (i < n) o[i] = 0.f;
}

// modes: 0 = mse(a,tgt); 1 = mse(sigmoid(a),tgt); 2 = mse(a, aux1-aux2); 3 = NLL over 15 logits/row
__global__ __launch_bounds__(256) void loss_kernel(
    const float* __restrict__ a, const float* __restrict__ tgt,
    const float* __restrict__ aux1, const float* __restrict__ aux2,
    const int* __restrict__ lab, int n, int mode, float w, float* __restrict__ out)
{
  int i = blockIdx.x * 256 + threadIdx.x;
  float v = 0.f;
  if (i < n) {
    if (mode == 0) { float d = a[i] - tgt[i]; v = d * d; }
    else if (mode == 1) { float s = 1.f / (1.f + __expf(-a[i])); float d = s - tgt[i]; v = d * d; }
    else if (mode == 2) { float u = aux1[i] - aux2[i]; float d = a[i] - u; v = d * d; }
    else {
      const float* lg = a + (size_t)i * 15;
      float mx = lg[0];
      for (int c = 1; c < 15; ++c) mx = fmaxf(mx, lg[c]);
      float s = 0.f;
      for (int c = 0; c < 15; ++c) s += __expf(lg[c] - mx);
      v = -(lg[lab[i]] - mx - logf(s));
    }
  }
  for (int o = 16; o > 0; o >>= 1) v += __shfl_xor(v, o, 32);
  __shared__ float ps[8];
  int lane = threadIdx.x & 31, wid = threadIdx.x >> 5;
  if (lane == 0) ps[wid] = v;
  __syncthreads();
  if (threadIdx.x == 0) {
    float s = 0.f;
    for (int k = 0; k < 8; ++k) s += ps[k];
    atomicAdd(out, s * w);
  }
}

// ---------------- host orchestration ----------------
struct Lin  { const float* b; const float* w; };
struct LNp  { const float* b; const float* g; };
struct MHAp { Lin k, o, q, v; };
struct Layerp { Lin ac1, ac2, at1, at2; MHAp ca; Lin ff1, ff2; LNp nc; MHAp sa; };

extern "C" void kernel_launch(void* const* d_in, const int* in_sizes, int n_in,
                              void* d_out, int out_size, void* d_ws, size_t ws_size,
                              hipStream_t stream)
{
  (void)in_sizes; (void)n_in; (void)out_size; (void)ws_size;
  const float* strokes      = (const float*)d_in[0];
  const float* canvas_dino  = (const float*)d_in[1];
  const float* clip_patch   = (const float*)d_in[2];
  const float* text_feat    = (const float*)d_in[3];
  const float* a_tar        = (const float*)d_in[4];
  const int*   label_tar    = (const int*)d_in[5];
  const float* normal_tar   = (const float*)d_in[6];
  const float* landmark_tar = (const float*)d_in[7];
  const float* dino_tar     = (const float*)d_in[8];
  const float* a_src        = (const float*)d_in[9];
  const float* t_rand       = (const float*)d_in[10];
  const float* past_k       = (const float*)d_in[11];
  const float* past_v       = (const float*)d_in[12];

  // params flattened in pytree (alphabetical) order; within a dense: b then w
  int pi = 13;
  auto F    = [&]() { return (const float*)d_in[pi++]; };
  auto LINf = [&]() { Lin l; l.b = F(); l.w = F(); return l; };
  auto LNf  = [&]() { LNp l; l.b = F(); l.g = F(); return l; };
  auto MHAf = [&]() { MHAp m; m.k = LINf(); m.o = LINf(); m.q = LINf(); m.v = LINf(); return m; };

  Lin canvas1 = LINf(), canvas2 = LINf(), clip_proj = LINf(), dinoL = LINf();
  Lin fm1 = LINf(), fm2 = LINf(), fm3 = LINf(), labelL = LINf(), landmarkL = LINf();
  Layerp ly[L_];
  for (int i = 0; i < L_; ++i) {
    ly[i].ac1 = LINf(); ly[i].ac2 = LINf(); ly[i].at1 = LINf(); ly[i].at2 = LINf();
    ly[i].ca = MHAf(); ly[i].ff1 = LINf(); ly[i].ff2 = LINf(); ly[i].nc = LNf(); ly[i].sa = MHAf();
  }
  Lin normalL = LINf();
  LNp outNorm = LNf();
  const float* pos_emb = F();
  Lin strokeP = LINf(), text1 = LINf(), text2 = LINf();

  // workspace bump allocator
  float* wsp = (float*)d_ws;
  auto alloc = [&](size_t n) { float* r = wsp; wsp += n; return r; };
  float* bufH  = alloc((size_t)B_ * 512);
  float* cCan  = alloc((size_t)B_ * D_);
  float* cText = alloc((size_t)B_ * D_);
  float* kvPat = alloc((size_t)B_ * P_ * D_);
  float* x     = alloc((size_t)B_ * T_ * D_);
  float* x2    = alloc((size_t)B_ * T_ * D_);
  float* scC   = alloc((size_t)B_ * 512);
  float* scT   = alloc((size_t)B_ * 512);
  float* kcat  = alloc((size_t)B_ * (PAST_ + T_) * D_);
  float* vcat  = alloc((size_t)B_ * (PAST_ + T_) * D_);
  float* qb    = alloc((size_t)B_ * T_ * D_);
  float* kb    = alloc((size_t)B_ * P_ * D_);
  float* vb    = alloc((size_t)B_ * P_ * D_);
  float* ab    = alloc((size_t)B_ * T_ * D_);
  float* ffh   = alloc((size_t)B_ * T_ * FF_);
  float* hb    = alloc((size_t)B_ * D_);
  float* fmin  = alloc((size_t)B_ * 296);
  float* g1    = alloc((size_t)B_ * 512);
  float* g2    = alloc((size_t)B_ * 512);
  float* pred  = alloc((size_t)B_ * 8);
  float* logit = alloc((size_t)B_ * 16);
  float* nrmO  = alloc((size_t)B_ * 4);
  float* lmkO  = alloc((size_t)B_ * 136);
  float* dinO  = alloc((size_t)B_ * 384);

  auto gemm = [&](const float* A, Lin l, const float* add, float* C, int M, int N, int K, int act) {
    dim3 g((N + GBN - 1) / GBN, (M + GBM - 1) / GBM);
    gemm_bf16_kernel<<<g, 256, 0, stream>>>(A, l.w, l.b, add, C, M, N, K, act);
  };
  auto ew = [&](int n) { return dim3((n + 255) / 256); };

  // conditioning MLPs + projections
  gemm(canvas_dino, canvas1, nullptr, bufH, B_, 512, 384, 1);
  gemm(bufH, canvas2, nullptr, cCan, B_, 256, 512, 0);
  gemm(text_feat, text1, nullptr, bufH, B_, 256, 512, 1);
  gemm(bufH, text2, nullptr, cText, B_, 256, 256, 0);
  gemm(clip_patch, clip_proj, nullptr, kvPat, B_ * P_, 256, 512, 0);
  gemm(strokes, strokeP, nullptr, x, B_ * T_, 256, 8, 0);
  add_pos_kernel<<<ew(B_ * T_ * D_), 256, 0, stream>>>(x, pos_emb, B_ * T_ * D_);

  for (int li = 0; li < L_; ++li) {
    const Layerp& lp = ly[li];
    gemm(cCan,  lp.ac1, nullptr, scC, B_, 512, 256, 0);
    gemm(cText, lp.at1, nullptr, scT, B_, 512, 256, 0);
    adaln_kernel<<<B_ * T_, 256, 0, stream>>>(x, scC, scT, x2);

    const float* pk = past_k + (size_t)li * B_ * PAST_ * D_;
    const float* pv = past_v + (size_t)li * B_ * PAST_ * D_;
    concat_kernel<<<ew(B_ * (PAST_ + T_) * D_), 256, 0, stream>>>(pk, x2, kcat, B_ * (PAST_ + T_) * D_);
    concat_kernel<<<ew(B_ * (PAST_ + T_) * D_), 256, 0, stream>>>(pv, x2, vcat, B_ * (PAST_ + T_) * D_);

    gemm(x2,   lp.sa.q, nullptr, qb, B_ * T_, 256, 256, 0);
    gemm(kcat, lp.sa.k, nullptr, kb, B_ * (PAST_ + T_), 256, 256, 0);
    gemm(vcat, lp.sa.v, nullptr, vb, B_ * (PAST_ + T_), 256, 256, 0);
    attn_kernel<<<B_ * H_, 256, 0, stream>>>(qb, kb, vb, ab, T_, PAST_ + T_, PAST_);
    gemm(ab, lp.sa.o, x, x, B_ * T_, 256, 256, 0);  // + residual

    ln_affine_kernel<<<B_ * T_, 256, 0, stream>>>(x, lp.nc.g, lp.nc.b, x2, 0, 1);
    gemm(x2,    lp.ca.q, nullptr, qb, B_ * T_, 256, 256, 0);
    gemm(kvPat, lp.ca.k, nullptr, kb, B_ * P_, 256, 256, 0);
    gemm(kvPat, lp.ca.v, nullptr, vb, B_ * P_, 256, 256, 0);
    attn_kernel<<<B_ * H_, 256, 0, stream>>>(qb, kb, vb, ab, T_, P_, -1);
    gemm(ab, lp.ca.o, x, x, B_ * T_, 256, 256, 0);  // + residual

    gemm(cCan,  lp.ac2, nullptr, scC, B_, 512, 256, 0);
    gemm(cText, lp.at2, nullptr, scT, B_, 512, 256, 0);
    adaln_kernel<<<B_ * T_, 256, 0, stream>>>(x, scC, scT, x2);
    gemm(x2,  lp.ff1, nullptr, ffh, B_ * T_, FF_, 256, 1);  // silu
    gemm(ffh, lp.ff2, x, x, B_ * T_, 256, FF_, 0);          // + residual
  }

  // heads: only the last token is needed; LN is rowwise so evaluate just row T-1
  ln_affine_kernel<<<B_, 256, 0, stream>>>(x, outNorm.g, outNorm.b, hb, T_ - 1, T_);
  fm_build_kernel<<<ew(B_ * 296), 256, 0, stream>>>(hb, t_rand, a_src, a_tar, fmin);
  gemm(fmin, fm1, nullptr, g1, B_, 512, 296, 1);
  gemm(g1, fm2, nullptr, g2, B_, 512, 512, 1);
  gemm(g2, fm3, nullptr, pred, B_, 8, 512, 0);
  gemm(hb, labelL,    nullptr, logit, B_, 15, 256, 0);
  gemm(hb, normalL,   nullptr, nrmO,  B_, 3, 256, 0);
  gemm(hb, landmarkL, nullptr, lmkO,  B_, 136, 256, 0);
  gemm(hb, dinoL,     nullptr, dinO,  B_, 384, 256, 0);

  float* out = (float*)d_out;
  zero_kernel<<<1, 256, 0, stream>>>(out, 1);
  loss_kernel<<<ew(B_ * 8), 256, 0, stream>>>(pred, nullptr, a_tar, a_src, nullptr,
                                              B_ * 8, 2, 1.0f / (B_ * 8), out);
  loss_kernel<<<ew(B_), 256, 0, stream>>>(logit, nullptr, nullptr, nullptr, label_tar,
                                          B_, 3, 0.1f / B_, out);
  loss_kernel<<<ew(B_ * 3), 256, 0, stream>>>(nrmO, normal_tar, nullptr, nullptr, nullptr,
                                              B_ * 3, 1, 0.1f / (B_ * 3), out);
  loss_kernel<<<ew(B_ * 136), 256, 0, stream>>>(lmkO, landmark_tar, nullptr, nullptr, nullptr,
                                                B_ * 136, 1, 0.1f / (B_ * 136), out);
  loss_kernel<<<ew(B_ * 384), 256, 0, stream>>>(dinO, dino_tar, nullptr, nullptr, nullptr,
                                                B_ * 384, 1, 0.1f / (B_ * 384), out);
}
